// BaseMLPExperts_16166256902678
// MI455X (gfx1250) — matrix operand order repaired and checked by reading, not compiled, not run on hardware
//
#include <hip/hip_runtime.h>
#include <cstdint>
#include <cstddef>

// ---------------------------------------------------------------------------
// MoE expert MLP, per expert e:
//   h1 = gelu_erf( x[:,e,:] @ wi[e] )   (8192x1024)@(1024x4096)
//   y[:,e,:] = h1 @ wo[e]               (8192x4096)@(4096x1024)
// bf16 WMMA (v_wmma_f32_16x16x32_bf16), fp32 accumulate.
// Double-buffered LDS pipeline; kernel2 A-tiles staged by the Tensor Data
// Mover (tensor_load_to_lds + s_wait_tensorcnt). Per-expert sequencing keeps
// the 64MB bf16 intermediate L2-resident.
// ---------------------------------------------------------------------------

#define T_DIM 8192
#define E_DIM 8
#define H_DIM 1024
#define I_DIM 4096

// LDS tile pitch in halfwords: 32 data + 8 pad -> 80B rows (bank-conflict-free,
// 8B aligned for ds_load_b64, and TDM-padding-compatible: 16DW data + 4DW pad).
#define LDS_PITCH 40

typedef __bf16 bf16x16 __attribute__((ext_vector_type(16)));
typedef float f32x8 __attribute__((ext_vector_type(8)));
typedef unsigned int v4u __attribute__((ext_vector_type(4)));
typedef int v8i_t __attribute__((ext_vector_type(8)));
typedef int v4i_t __attribute__((ext_vector_type(4)));

#if defined(__has_builtin)
#if __has_builtin(__builtin_amdgcn_tensor_load_to_lds) && __has_builtin(__builtin_amdgcn_s_wait_tensorcnt)
#define USE_TDM 1
#endif
#endif

union FragU {
    unsigned int u[8];
    bf16x16 v;
};

// fp32 -> bf16, round-half-up (1 VALU), value in bits [31:16] so the
// compiler can use ds_store_b16_d16_hi for LDS scatter stores.
__device__ __forceinline__ unsigned int f2bf_hi(float f) {
    return __float_as_uint(f) + 0x8000u;
}
__device__ __forceinline__ unsigned short f2bf(float f) {
    return (unsigned short)(f2bf_hi(f) >> 16);
}
// Pack two bf16 with a single v_perm_b32: D = {hi[3],hi[2],lo[3],lo[2]}
__device__ __forceinline__ unsigned int pack2bf(float a, float b) {
    return __builtin_amdgcn_perm(f2bf_hi(b), f2bf_hi(a), 0x07060302u);
}

__device__ __forceinline__ float gelu_erf(float x) {
    return 0.5f * x * (1.0f + erff(x * 0.70710678118654752f));
}

// ---- A-fragment (16x32 bf16, ISA 7.12.2 layout) from LDS tile [row][k] ----
__device__ __forceinline__ bf16x16 load_a_frag(const unsigned short* As, int rowBase, int lane) {
    FragU f;
    const int m = rowBase + (lane & 15);
    const int kh = (lane >> 4) * 8;
    const uint2* lo = (const uint2*)&As[m * LDS_PITCH + kh];
    const uint2* hi = (const uint2*)&As[m * LDS_PITCH + 16 + kh];
    uint2 a0 = lo[0], a1 = lo[1];
    uint2 a2 = hi[0], a3 = hi[1];
    f.u[0] = a0.x; f.u[1] = a0.y; f.u[2] = a1.x; f.u[3] = a1.y;
    f.u[4] = a2.x; f.u[5] = a2.y; f.u[6] = a3.x; f.u[7] = a3.y;
    return f.v;
}

// ---- B-fragment (32x16 bf16) from LDS tile stored transposed [n][k] ----
__device__ __forceinline__ bf16x16 load_b_frag(const unsigned short* Bs, int colBase, int lane) {
    FragU f;
    const int n = colBase + (lane & 15);
    const int kb = (lane >> 4) * 8;
    const uint2* p = (const uint2*)&Bs[n * LDS_PITCH + kb];
    const uint2* q = (const uint2*)&Bs[n * LDS_PITCH + 16 + kb];
    uint2 b0 = p[0], b1 = p[1], b2 = q[0], b3 = q[1];
    f.u[0] = b0.x; f.u[1] = b0.y; f.u[2] = b1.x; f.u[3] = b1.y;
    f.u[4] = b2.x; f.u[5] = b2.y; f.u[6] = b3.x; f.u[7] = b3.y;
    return f.v;
}

// ---- 8x v_wmma_f32_16x16x32_bf16 on one staged K-slice ----
__device__ __forceinline__ void mma_tile(const unsigned short* As, const unsigned short* Bs,
                                         int warpM, int warpN, int lane, f32x8 (&acc)[4][2]) {
    bf16x16 af[4], bfr[2];
#pragma unroll
    for (int i = 0; i < 4; ++i) af[i] = load_a_frag(As, warpM * 64 + i * 16, lane);
#pragma unroll
    for (int j = 0; j < 2; ++j) bfr[j] = load_b_frag(Bs, warpN * 32 + j * 16, lane);
#pragma unroll
    for (int i = 0; i < 4; ++i)
#pragma unroll
        for (int j = 0; j < 2; ++j)
            acc[i][j] = __builtin_amdgcn_wmma_f32_16x16x32_bf16(
                false, af[i], false, bfr[j], (short)0, acc[i][j], false, false);
}

#if defined(USE_TDM)
// ---- Tensor Data Mover: 2D bf16 tile (128 rows x 32 cols) -> padded LDS ----
// D# per CDNA5 ISA: group0 = {count, lds_addr,
// global_addr, type=2}; group1 = {data_size=2B, pad 16DW->+4DW, dims/strides}.
// 6-arg builtin form (clang-23): (v4u g0, v8i g1, v4i g2, v4i g3, v8i, i32 cpol)
__device__ __forceinline__ void tdm_load_a_tile(const unsigned short* gsrc, unsigned lds_off,
                                                unsigned rowStrideElems) {
    const unsigned long long ga = (unsigned long long)(uintptr_t)gsrc;
    v4u g0;
    g0.x = 1u;                                             // count=1 valid descriptor
    g0.y = lds_off;                                        // LDS byte address
    g0.z = (unsigned)(ga & 0xFFFFFFFFu);                   // global_addr[31:0]
    g0.w = (unsigned)((ga >> 32) & 0x1FFFFFFu) | (2u << 30); // addr[56:32] | type=2
    v8i_t g1;
    g1[0] = (1 << 16)        // data_size = 2 bytes
          | (1 << 20)        // pad_enable
          | (3 << 22)        // pad_interval: every 16 DWORDs (= 64B data row)
          | (3 << 25);       // pad_amount: 4 DWORDs (16B) -> 80B LDS pitch
    g1[1] = (int)((rowStrideElems & 0xFFFFu) << 16);       // tensor_dim0 lo16
    g1[2] = (int)((rowStrideElems >> 16) & 0xFFFFu)
          | (int)((T_DIM & 0xFFFFu) << 16);                // tensor_dim0 hi | tensor_dim1 lo
    g1[3] = (int)((T_DIM >> 16) & 0xFFFFu) | (32 << 16);   // tensor_dim1 hi | tile_dim0=32
    g1[4] = 128;                                           // tile_dim1=128, tile_dim2=0
    g1[5] = (int)rowStrideElems;                           // tensor_dim0_stride lo32
    g1[6] = 0;
    g1[7] = 0;
    v4i_t z4 = {0, 0, 0, 0};
    v8i_t z8 = {0, 0, 0, 0, 0, 0, 0, 0};
    __builtin_amdgcn_tensor_load_to_lds(g0, g1, z4, z4, z8, 0);
}
#endif

// ===========================================================================
// Kernel 1: h1 = gelu(x_e @ wi_e)  M=8192 N=4096 K=1024, double-buffered LDS
// ===========================================================================
__global__ __launch_bounds__(256)
void moe_gemm1_gelu(const float* __restrict__ x, const float* __restrict__ wi,
                    unsigned short* __restrict__ h1, int e) {
    __shared__ __align__(16) unsigned short As[2][128 * LDS_PITCH];
    __shared__ __align__(16) unsigned short Bs[2][128 * LDS_PITCH];

    const int tid   = threadIdx.x;
    const int lane  = tid & 31;
    const int wave  = tid >> 5;
    const int warpM = wave >> 2;
    const int warpN = wave & 3;
    const int bm = blockIdx.y * 128;
    const int bn = blockIdx.x * 128;

    const float* xe  = x  + (size_t)e * H_DIM;
    const float* wie = wi + (size_t)e * ((size_t)H_DIM * I_DIM);

    const int ar = tid >> 3;          // A: 32 rows/pass, 4 passes
    const int ac = (tid & 7) << 2;
    const int br = tid >> 5;          // B: 8 k-rows/pass, 4 passes
    const int bc = (tid & 31) << 2;

    f32x8 acc[4][2];
#pragma unroll
    for (int i = 0; i < 4; ++i)
#pragma unroll
        for (int j = 0; j < 2; ++j)
#pragma unroll
            for (int r = 0; r < 8; ++r)
                acc[i][j][r] = 0.0f;

    float4 aR[4], bR[4];
    // ---- prologue: stage K-tile 0 into buffer 0
#pragma unroll
    for (int i = 0; i < 4; ++i)
        aR[i] = *(const float4*)(xe + (size_t)(bm + ar + 32 * i) * (E_DIM * H_DIM) + ac);
#pragma unroll
    for (int i = 0; i < 4; ++i)
        bR[i] = *(const float4*)(wie + (size_t)(br + 8 * i) * I_DIM + (bn + bc));
#pragma unroll
    for (int i = 0; i < 4; ++i) {
        uint2 p; p.x = pack2bf(aR[i].x, aR[i].y); p.y = pack2bf(aR[i].z, aR[i].w);
        *(uint2*)&As[0][(ar + 32 * i) * LDS_PITCH + ac] = p;
    }
#pragma unroll
    for (int i = 0; i < 4; ++i) {
        const int r = br + 8 * i;
        Bs[0][(bc + 0) * LDS_PITCH + r] = f2bf(bR[i].x);
        Bs[0][(bc + 1) * LDS_PITCH + r] = f2bf(bR[i].y);
        Bs[0][(bc + 2) * LDS_PITCH + r] = f2bf(bR[i].z);
        Bs[0][(bc + 3) * LDS_PITCH + r] = f2bf(bR[i].w);
    }
    __syncthreads();

    for (int k0 = 0, it = 0; k0 < H_DIM; k0 += 32, ++it) {
        const int cur = it & 1, nxt = cur ^ 1;
        const bool hasNext = (k0 + 32) < H_DIM;
        if (hasNext) {
            const int kn = k0 + 32;
#pragma unroll
            for (int i = 0; i < 4; ++i)
                aR[i] = *(const float4*)(xe + (size_t)(bm + ar + 32 * i) * (E_DIM * H_DIM) + (kn + ac));
#pragma unroll
            for (int i = 0; i < 4; ++i)
                bR[i] = *(const float4*)(wie + (size_t)(kn + br + 8 * i) * I_DIM + (bn + bc));
            if (k0 + 64 < H_DIM) {   // deep prefetch (global_prefetch_b8)
                __builtin_prefetch(xe + (size_t)(bm + ar) * (E_DIM * H_DIM) + (k0 + 64 + ac), 0, 1);
                __builtin_prefetch(wie + (size_t)(k0 + 64 + br) * I_DIM + (bn + bc), 0, 1);
            }
        }

        mma_tile(As[cur], Bs[cur], warpM, warpN, lane, acc);

        if (hasNext) {
#pragma unroll
            for (int i = 0; i < 4; ++i) {
                uint2 p; p.x = pack2bf(aR[i].x, aR[i].y); p.y = pack2bf(aR[i].z, aR[i].w);
                *(uint2*)&As[nxt][(ar + 32 * i) * LDS_PITCH + ac] = p;
            }
#pragma unroll
            for (int i = 0; i < 4; ++i) {
                const int r = br + 8 * i;
                Bs[nxt][(bc + 0) * LDS_PITCH + r] = f2bf(bR[i].x);
                Bs[nxt][(bc + 1) * LDS_PITCH + r] = f2bf(bR[i].y);
                Bs[nxt][(bc + 2) * LDS_PITCH + r] = f2bf(bR[i].z);
                Bs[nxt][(bc + 3) * LDS_PITCH + r] = f2bf(bR[i].w);
            }
        }
        __syncthreads();
    }

    // ---- epilogue: exact-erf GELU, bf16 store to h1
    const int rOff = (lane >> 4) * 8;
    const int cOff = (lane & 15);
#pragma unroll
    for (int i = 0; i < 4; ++i) {
        const int row0 = bm + warpM * 64 + i * 16 + rOff;
#pragma unroll
        for (int j = 0; j < 2; ++j) {
            const int col = bn + warpN * 32 + j * 16 + cOff;
#pragma unroll
            for (int r = 0; r < 8; ++r)
                h1[(size_t)(row0 + r) * I_DIM + col] = f2bf(gelu_erf(acc[i][j][r]));
        }
    }
}

// ===========================================================================
// Kernel 2: y = h1 @ wo_e  M=8192 N=1024 K=4096; A-tiles via TDM DMA
// ===========================================================================
__global__ __launch_bounds__(256)
void moe_gemm2(const unsigned short* __restrict__ h1, const float* __restrict__ wo,
               float* __restrict__ out, int e) {
    __shared__ __align__(16) unsigned short As[2][128 * LDS_PITCH];
    __shared__ __align__(16) unsigned short Bs[2][128 * LDS_PITCH];

    const int tid   = threadIdx.x;
    const int lane  = tid & 31;
    const int wave  = tid >> 5;
    const int warpM = wave >> 2;
    const int warpN = wave & 3;
    const int bm = blockIdx.y * 128;
    const int bn = blockIdx.x * 128;

    const float* woe = wo + (size_t)e * ((size_t)I_DIM * H_DIM);

    const int ar = tid >> 3;
    const int ac = (tid & 7) << 2;
    const int br = tid >> 5;
    const int bc = (tid & 31) << 2;

    f32x8 acc[4][2];
#pragma unroll
    for (int i = 0; i < 4; ++i)
#pragma unroll
        for (int j = 0; j < 2; ++j)
#pragma unroll
            for (int r = 0; r < 8; ++r)
                acc[i][j][r] = 0.0f;

    float4 bR[4];
#if !defined(USE_TDM)
    uint2 aR[4];
#endif

    // ---- prologue: stage K-tile 0 into buffer 0
#if defined(USE_TDM)
    if (wave == 0)
        tdm_load_a_tile(h1 + (size_t)bm * I_DIM, (unsigned)(uintptr_t)&As[0][0], I_DIM);
#else
#pragma unroll
    for (int i = 0; i < 4; ++i)
        aR[i] = *(const uint2*)(h1 + (size_t)(bm + ar + 32 * i) * I_DIM + ac);
#pragma unroll
    for (int i = 0; i < 4; ++i)
        *(uint2*)&As[0][(ar + 32 * i) * LDS_PITCH + ac] = aR[i];
#endif
#pragma unroll
    for (int i = 0; i < 4; ++i)
        bR[i] = *(const float4*)(woe + (size_t)(br + 8 * i) * H_DIM + (bn + bc));
#pragma unroll
    for (int i = 0; i < 4; ++i) {
        const int r = br + 8 * i;
        Bs[0][(bc + 0) * LDS_PITCH + r] = f2bf(bR[i].x);
        Bs[0][(bc + 1) * LDS_PITCH + r] = f2bf(bR[i].y);
        Bs[0][(bc + 2) * LDS_PITCH + r] = f2bf(bR[i].z);
        Bs[0][(bc + 3) * LDS_PITCH + r] = f2bf(bR[i].w);
    }
#if defined(USE_TDM)
    if (wave == 0) __builtin_amdgcn_s_wait_tensorcnt(0);
#endif
    __syncthreads();

    for (int k0 = 0, it = 0; k0 < I_DIM; k0 += 32, ++it) {
        const int cur = it & 1, nxt = cur ^ 1;
        const bool hasNext = (k0 + 32) < I_DIM;
        if (hasNext) {
            const int kn = k0 + 32;
#if defined(USE_TDM)
            if (wave == 0)
                tdm_load_a_tile(h1 + (size_t)bm * I_DIM + kn,
                                (unsigned)(uintptr_t)&As[nxt][0], I_DIM);
#else
#pragma unroll
            for (int i = 0; i < 4; ++i)
                aR[i] = *(const uint2*)(h1 + (size_t)(bm + ar + 32 * i) * I_DIM + (kn + ac));
#endif
#pragma unroll
            for (int i = 0; i < 4; ++i)
                bR[i] = *(const float4*)(woe + (size_t)(kn + br + 8 * i) * H_DIM + (bn + bc));
        }

        mma_tile(As[cur], Bs[cur], warpM, warpN, lane, acc);

        if (hasNext) {
#if !defined(USE_TDM)
#pragma unroll
            for (int i = 0; i < 4; ++i)
                *(uint2*)&As[nxt][(ar + 32 * i) * LDS_PITCH + ac] = aR[i];
#endif
#pragma unroll
            for (int i = 0; i < 4; ++i) {
                const int r = br + 8 * i;
                Bs[nxt][(bc + 0) * LDS_PITCH + r] = f2bf(bR[i].x);
                Bs[nxt][(bc + 1) * LDS_PITCH + r] = f2bf(bR[i].y);
                Bs[nxt][(bc + 2) * LDS_PITCH + r] = f2bf(bR[i].z);
                Bs[nxt][(bc + 3) * LDS_PITCH + r] = f2bf(bR[i].w);
            }
#if defined(USE_TDM)
            if (wave == 0) __builtin_amdgcn_s_wait_tensorcnt(0);
#endif
        }
        __syncthreads();
    }

    // ---- epilogue: fp32 store into interleaved [T][E][H]
    const int rOff = (lane >> 4) * 8;
    const int cOff = (lane & 15);
#pragma unroll
    for (int i = 0; i < 4; ++i) {
        const int row0 = bm + warpM * 64 + i * 16 + rOff;
#pragma unroll
        for (int j = 0; j < 2; ++j) {
            const int col = bn + warpN * 32 + j * 16 + cOff;
#pragma unroll
            for (int r = 0; r < 8; ++r)
                out[(size_t)(row0 + r) * (E_DIM * H_DIM) + (size_t)e * H_DIM + col] = acc[i][j][r];
        }
    }
}

// ===========================================================================
extern "C" void kernel_launch(void* const* d_in, const int* in_sizes, int n_in,
                              void* d_out, int out_size, void* d_ws, size_t ws_size,
                              hipStream_t stream) {
    (void)in_sizes; (void)n_in; (void)out_size; (void)ws_size;
    const float* x  = (const float*)d_in[0];
    const float* wi = (const float*)d_in[1];
    const float* wo = (const float*)d_in[2];
    float* out = (float*)d_out;
    unsigned short* h1 = (unsigned short*)d_ws;   // [8192 x 4096] bf16, 64MB, reused per expert

    dim3 block(256);
    dim3 grid1(I_DIM / 128, T_DIM / 128);   // 32 x 64
    dim3 grid2(H_DIM / 128, T_DIM / 128);   //  8 x 64

    for (int e = 0; e < E_DIM; ++e) {
        moe_gemm1_gelu<<<grid1, block, 0, stream>>>(x, wi, h1, e);
        moe_gemm2<<<grid2, block, 0, stream>>>(h1, wo, out, e);
    }
}
